// ConstrainedMLP_24635932410177
// MI455X (gfx1250) — matrix operand order repaired and checked by reading, not compiled, not run on hardware
//
#include <hip/hip_runtime.h>
#include <hip/hip_bf16.h>

// ---------------------------------------------------------------------------
// ConstrainedMLP on gfx1250 (MI455X): f16 WMMA pipeline, v2.
//   B=262144, IN=12, HID=256, OUT=12.
//   prep kernel: f32 weights -> fragment-major f16 in d_ws (contiguous 1KB
//                fragments => perfectly coalesced 32B/lane global loads).
//   main kernel: 128 rows/block, 8 waves; W1+W2 fragments register-resident,
//                loaded straight from L2 (no LDS staging pass). LDS holds only
//                the inter-layer activations in A-fragment layout: 136 KB =>
//                two workgroups per WGP (4 waves/SIMD).
// ---------------------------------------------------------------------------

typedef __attribute__((ext_vector_type(16))) _Float16 v16h;
typedef __attribute__((ext_vector_type(8)))  float    v8f;

#define WMMA_F16(a, b, c) \
    __builtin_amdgcn_wmma_f32_16x16x32_f16(false, (a), false, (b), (short)0, (c), false, false)

// ---- workspace layout (in _Float16 units) ----
// W1B: 16 ntiles * 32 lanes * 16 halves          = 8192   (K padded 12->32)
// W2B: 8 ktiles * 16 ntiles * 32 * 16            = 65536
// W3B: 8 ktiles * 1 ntile  * 32 * 16             = 4096   (N padded 12->16)
#define W1B_OFF   0
#define W2B_OFF   8192
#define W3B_OFF   73728
#define WTOT      77824            // halves (155648 bytes of d_ws)

// ---- rows per block ----
#define RT        8                // 8 row-tiles of 16 = 128 rows/block
#define NBLOCKS   2048             // 262144 / 128

// ---- dynamic LDS layout ----
// sH1: [RT][8 kt][32 lane][16 halves] = 32768 halves = 64 KB
// sH2: same = 64 KB
// sX : [128 rows][16 cols] f32 = 8 KB
#define SH1_HALVES (RT * 8 * 32 * 16)
#define SX_BYTE    (2 * SH1_HALVES * 2)
#define SMEM_BYTES (SX_BYTE + RT * 16 * 16 * 4)   // 139264 B (~136 KB); 2 WGs/WGP

// ===========================================================================
// Prep: convert weights f32 -> f16 in B-fragment-major order.
// B 32x16 f16 fragment: lane<16 holds col=lane, K = e (e<8), 16+(e-8) (e>=8);
//                       lane>=16 holds col=lane-16, K = 8+e (e<8), 24+(e-8) (e>=8).
// ===========================================================================
__global__ __launch_bounds__(256) void mlp_prep(const float* __restrict__ W1,
                                                const float* __restrict__ W2,
                                                const float* __restrict__ W3,
                                                _Float16* __restrict__ ws) {
    int idx = blockIdx.x * 256 + threadIdx.x;
    if (idx < W2B_OFF) {                         // W1B, K padded to 32
        int e = idx & 15, lane = (idx >> 4) & 31, nt = idx >> 9;
        int k   = (e & 7) + ((e & 8) << 1) + ((lane >= 16) ? 8 : 0);
        int col = nt * 16 + (lane & 15);
        float v = (k < 12) ? W1[k * 256 + col] : 0.0f;
        ws[idx] = (_Float16)v;
    } else if (idx < W3B_OFF) {                  // W2B
        int i = idx - W2B_OFF;
        int e = i & 15, lane = (i >> 4) & 31, nt = (i >> 9) & 15, kt = i >> 13;
        int k   = kt * 32 + (e & 7) + ((e & 8) << 1) + ((lane >= 16) ? 8 : 0);
        int col = nt * 16 + (lane & 15);
        ws[idx] = (_Float16)W2[k * 256 + col];
    } else if (idx < WTOT) {                     // W3B, N padded to 16
        int i = idx - W3B_OFF;
        int e = i & 15, lane = (i >> 4) & 31, kt = i >> 9;
        int k   = kt * 32 + (e & 7) + ((e & 8) << 1) + ((lane >= 16) ? 8 : 0);
        int col = lane & 15;
        float v = (col < 12) ? W3[k * 12 + col] : 0.0f;
        ws[idx] = (_Float16)v;
    }
}

// Inverse of the 16-bit 16x32 A-fragment map: for local K index kl (0..31),
// the element lives in lane half (kl>>3)&1 at half-index e=(kl&7)|((kl&16)>>1).
__device__ __forceinline__ void store_Afrag_scatter(_Float16* __restrict__ sH,
                                                    int rt, int K, int hi,
                                                    const v8f& acc, bool relu) {
    int kt2       = K >> 5;
    int kl        = K & 31;
    int e         = (kl & 7) | ((kl & 16) >> 1);
    int laneHiBit = (kl >> 3) & 1;
#pragma unroll
    for (int g = 0; g < 8; ++g) {
        int rowl     = g + 8 * hi;               // row within 16-row tile (D layout)
        int destLane = rowl + 16 * laneHiBit;
        float v = acc[g];
        if (relu) v = fmaxf(v, 0.0f);
        sH[((rt * 8 + kt2) * 32 + destLane) * 16 + e] = (_Float16)v;
    }
}

// ===========================================================================
// Main kernel: block = 256 threads (8 waves), 128 batch rows.
// ===========================================================================
__global__ __launch_bounds__(256) void mlp_main(const float* __restrict__ inp,
                                                const float* __restrict__ b1,
                                                const float* __restrict__ b2,
                                                const float* __restrict__ b3,
                                                const _Float16* __restrict__ ws,
                                                float* __restrict__ out) {
    extern __shared__ char smem[];
    _Float16* sH1 = (_Float16*)smem;                  // [RT][8 kt][32 lane][16 e]
    _Float16* sH2 = sH1 + SH1_HALVES;
    float*    sX  = (float*)(smem + SX_BYTE);         // [128 rows][16 cols]

    const int  tid   = threadIdx.x;
    const int  lane  = tid & 31;
    const int  wv    = tid >> 5;
    const int  colLo = lane & 15;
    const int  hi    = (lane >= 16) ? 1 : 0;
    const long rowBase = (long)blockIdx.x * (RT * 16);

    // ---- weights straight from global (L2-resident, 32B/lane coalesced) ----
    // This wave owns ntiles {2wv, 2wv+1} of HID.
    v16h W1f[2];
#pragma unroll
    for (int j = 0; j < 2; ++j) {
        int nt = 2 * wv + j;
        W1f[j] = *(const v16h*)&ws[W1B_OFF + (nt * 32 + lane) * 16];
    }
    v16h B2f[16];
#pragma unroll
    for (int kt = 0; kt < 8; ++kt)
#pragma unroll
        for (int j = 0; j < 2; ++j) {
            int nt = 2 * wv + j;
            B2f[kt * 2 + j] = *(const v16h*)&ws[W2B_OFF + ((kt * 16 + nt) * 32 + lane) * 16];
        }
    float bb0 = b1[(2 * wv + 0) * 16 + colLo];
    float bb1 = b1[(2 * wv + 1) * 16 + colLo];
    float cb0 = b2[(2 * wv + 0) * 16 + colLo];
    float cb1 = b2[(2 * wv + 1) * 16 + colLo];

    // =================== Layer 1: [16x32pad] @ [32x256] ===================
#pragma unroll
    for (int rt = 0; rt < RT; ++rt) {
        long r = rowBase + rt * 16 + colLo;
        const float* rp = inp + r * 12;
        v16h a1;
#pragma unroll
        for (int i = 0; i < 16; ++i) a1[i] = (_Float16)0.0f;
        if (!hi) {                                // lane<16: K=0..7 in e0..7
            float4 q0 = ((const float4*)rp)[0];
            float4 q1 = ((const float4*)rp)[1];
            a1[0] = (_Float16)q0.x; a1[1] = (_Float16)q0.y;
            a1[2] = (_Float16)q0.z; a1[3] = (_Float16)q0.w;
            a1[4] = (_Float16)q1.x; a1[5] = (_Float16)q1.y;
            a1[6] = (_Float16)q1.z; a1[7] = (_Float16)q1.w;
        } else {                                  // lane>=16: K=8..11 in e0..3
            float4 q2 = ((const float4*)rp)[2];
            a1[0] = (_Float16)q2.x; a1[1] = (_Float16)q2.y;
            a1[2] = (_Float16)q2.z; a1[3] = (_Float16)q2.w;
        }
#pragma unroll
        for (int j = 0; j < 2; ++j) {
            float bias = j ? bb1 : bb0;
            v8f acc;
#pragma unroll
            for (int g = 0; g < 8; ++g) acc[g] = bias;
            acc = WMMA_F16(a1, W1f[j], acc);
            store_Afrag_scatter(sH1, rt, (2 * wv + j) * 16 + colLo, hi, acc, true);
        }
    }
    __syncthreads();

    // =================== Layer 2: [16x256] @ [256x256] ====================
#pragma unroll
    for (int rt = 0; rt < RT; ++rt) {
        v8f acc0, acc1;
#pragma unroll
        for (int g = 0; g < 8; ++g) { acc0[g] = cb0; acc1[g] = cb1; }
#pragma unroll
        for (int kt = 0; kt < 8; ++kt) {
            v16h a = *(const v16h*)&sH1[((rt * 8 + kt) * 32 + lane) * 16];
            acc0 = WMMA_F16(a, B2f[kt * 2 + 0], acc0);
            acc1 = WMMA_F16(a, B2f[kt * 2 + 1], acc1);
        }
        store_Afrag_scatter(sH2, rt, (2 * wv + 0) * 16 + colLo, hi, acc0, true);
        store_Afrag_scatter(sH2, rt, (2 * wv + 1) * 16 + colLo, hi, acc1, true);
    }
    __syncthreads();

    // =================== Layer 3: [16x256] @ [256x16pad] ==================
    {
        int rt = wv;                              // all 8 waves, one row-tile each
        float bias3 = (colLo < 12) ? b3[colLo] : 0.0f;
        v8f acc;
#pragma unroll
        for (int g = 0; g < 8; ++g) acc[g] = bias3;
#pragma unroll
        for (int kt = 0; kt < 8; ++kt) {
            v16h a  = *(const v16h*)&sH2[((rt * 8 + kt) * 32 + lane) * 16];
            v16h bf = *(const v16h*)&ws[W3B_OFF + (kt * 32 + lane) * 16];
            acc = WMMA_F16(a, bf, acc);
        }
#pragma unroll
        for (int g = 0; g < 8; ++g)              // D layout -> row-major x tile
            sX[(rt * 16 + g + 8 * hi) * 16 + colLo] = acc[g];
    }
    __syncthreads();

    // =================== Constraint post-processing (f32 VALU) ============
    if (tid < RT * 16) {                          // 128 rows, one lane each
        long R = rowBase + tid;
        const float* xr = &sX[tid * 16];
        const float* ir = inp + R * 12;
        float in0 = ir[0], in1 = ir[1], in2 = ir[2];
        float in6 = ir[6], in7 = ir[7], in8 = ir[8], in9 = ir[9];

        float p0 = tanhf(xr[0]), p1 = tanhf(xr[1]), p2 = tanhf(xr[2]);
        float v0 = tanhf(xr[3]), v1 = tanhf(xr[4]), v2 = tanhf(xr[5]);
        float pts = 1.0f / (1.0f + expf(-xr[6]));
        float c0 = tanhf(xr[7]), c1 = tanhf(xr[8]), c2 = tanhf(xr[9]);
        float s0 = tanhf(xr[10]), s1 = tanhf(xr[11]);

        // pos: project onto unit ball
        float dist = sqrtf(p0 * p0 + p1 * p1 + p2 * p2);
        float pinv = 1.0f / dist;
        float pc0 = (dist > 1.0f) ? p0 * pinv : p0;
        float pc1 = (dist > 1.0f) ? p1 * pinv : p1;
        float pc2 = (dist > 1.0f) ? p2 * pinv : p2;

        // pts: clamp hi at 1, lo at previous value
        float ptsc = (pts > 1.0f) ? 1.0f : pts;
        ptsc = (pts < in6) ? in6 : ptsc;

        // clus: deputy-offset + unit-ball projection
        float d0 = c0 - in7, d1 = c1 - in8, d2 = c2 - in9;
        float dp0 = in0 - in7, dp1 = in1 - in8, dp2 = in2 - in9;
        float dinv = 1.0f / sqrtf(dp0 * dp0 + dp1 * dp1 + dp2 * dp2);
        float dd = d0 * dp0 + d1 * dp1 + d2 * dp2;
        float o0 = (dd > 0.0f) ? c0 - dp0 * dinv : c0;
        float o1 = (dd > 0.0f) ? c1 - dp1 * dinv : c1;
        float o2 = (dd > 0.0f) ? c2 - dp2 * dinv : c2;
        float cd = sqrtf(o0 * o0 + o1 * o1 + o2 * o2);
        float cinv = 1.0f / cd;
        float cc0 = (cd > 1.0f) ? o0 * cinv : c0;
        float cc1 = (cd > 1.0f) ? o1 * cinv : c1;
        float cc2 = (cd > 1.0f) ? o2 * cinv : c2;

        // sun: normalize
        float sinv = 1.0f / sqrtf(s0 * s0 + s1 * s1);
        float su0 = s0 * sinv, su1 = s1 * sinv;

        float4* op = (float4*)(out + R * 12);    // 48B-aligned (R*12*4)
        op[0] = make_float4(pc0, pc1, pc2, v0);
        op[1] = make_float4(v1, v2, ptsc, cc0);
        op[2] = make_float4(cc1, cc2, su0, su1);
    }
}

// ===========================================================================
extern "C" void kernel_launch(void* const* d_in, const int* in_sizes, int n_in,
                              void* d_out, int out_size, void* d_ws, size_t ws_size,
                              hipStream_t stream) {
    const float* inp = (const float*)d_in[0];
    const float* W1  = (const float*)d_in[1];
    const float* b1  = (const float*)d_in[2];
    const float* W2  = (const float*)d_in[3];
    const float* b2  = (const float*)d_in[4];
    const float* W3  = (const float*)d_in[5];
    const float* b3  = (const float*)d_in[6];
    _Float16*    ws  = (_Float16*)d_ws;          // needs 155648 bytes of d_ws
    float*       out = (float*)d_out;

    mlp_prep<<<WTOT / 256, 256, 0, stream>>>(W1, W2, W3, ws);
    mlp_main<<<NBLOCKS, 256, SMEM_BYTES, stream>>>(inp, b1, b2, b3, ws, out);
}
